// DAPEAttention_28776280883494
// MI455X (gfx1250) — matrix operand order repaired and checked by reading, not compiled, not run on hardware
//
#include <hip/hip_runtime.h>

#define B_   2
#define L_   1024
#define H_   16
#define D_   64
#define HID_ 1024

typedef __attribute__((ext_vector_type(16))) __bf16 v16bf;
typedef __attribute__((ext_vector_type(8)))  __bf16 v8bf;
typedef __attribute__((ext_vector_type(8)))  float  v8f;
typedef __attribute__((ext_vector_type(8)))  float  v8fv;

static __device__ inline v8f v8f_zero() {
  v8f z;
#pragma unroll
  for (int i = 0; i < 8; ++i) z[i] = 0.0f;
  return z;
}

// ---------------------------------------------------------------------------
// CDNA5 async global->LDS copy (ASYNCcnt path), CDNA5 ISA
// ---------------------------------------------------------------------------
static __device__ inline void async_load_b128(void* lds_dst, const void* gsrc) {
  unsigned ldsoff = (unsigned)(size_t)lds_dst;   // flat LDS addr truncates to LDS offset
  asm volatile("global_load_async_to_lds_b128 %0, %1, off"
               :: "v"(ldsoff), "v"(gsrc)
               : "memory");
}
static __device__ inline void wait_async0() {
  asm volatile("s_wait_asynccnt 0x0" ::: "memory");
}
// drain current tile's 4 loads while next tile's 4 stay in flight
static __device__ inline void wait_async4() {
  asm volatile("s_wait_asynccnt 0x4" ::: "memory");
}

// ---------------------------------------------------------------------------
// WMMA fragment loaders (CDNA5 wave32 layouts, CDNA5 ISA §7.12.2)
// ---------------------------------------------------------------------------

// A-matrix 16x32 bf16, row-major source, leading dim `ld` (elements).
static __device__ inline v16bf load_frag_a(const __bf16* p, int ld) {
  const int l  = threadIdx.x & 31;
  const int m  = l & 15;
  const int hf = l >> 4;
  const __bf16* r = p + (size_t)m * ld;
  v16bf out;
#pragma unroll
  for (int e = 0; e < 16; ++e) out[e] = r[hf * 16 + e];
  return out;
}

// B-matrix 32x16 bf16 where source holds B^T row-major (N rows of K elems).
static __device__ inline v16bf load_frag_bT(const __bf16* p, int ld) {
  const int l  = threadIdx.x & 31;
  const int n  = l & 15;
  const int hf = l >> 4;
  const __bf16* r = p + (size_t)n * ld + hf * 16;
  v16bf out;
#pragma unroll
  for (int e = 0; e < 16; ++e) out[e] = r[e];
  return out;
}

// B-matrix 32x16 bf16 from a row-major KxN source (strided; loop-invariant use only).
static __device__ inline v16bf load_frag_b_rm(const __bf16* p, int ld) {
  const int l  = threadIdx.x & 31;
  const int n  = l & 15;
  const int hf = l >> 4;
  v16bf out;
#pragma unroll
  for (int e = 0; e < 16; ++e) out[e] = p[(size_t)(hf * 16 + e) * ld + n];
  return out;
}

// B-matrix 32x16 where only K=0..15 is real; source is B^T row-major
// (N rows x >=16 K elems, leading dim ld).  K=16..31 lanes get zeros.
static __device__ inline v16bf load_frag_bT_k16(const __bf16* p, int ld) {
  const int l  = threadIdx.x & 31;
  const int n  = l & 15;
  const int hf = l >> 4;
  v16bf out;
  if (hf == 0) {
    const __bf16* r = p + (size_t)n * ld;
#pragma unroll
    for (int e = 0; e < 16; ++e) out[e] = r[e];
  } else {
#pragma unroll
    for (int e = 0; e < 16; ++e) out[e] = (__bf16)0.0f;
  }
  return out;
}

// ---------------------------------------------------------------------------
// Kernel 1a: f32 -> bf16 cast, 8 elements / thread (b128 in, b128 out)
// ---------------------------------------------------------------------------
__global__ void cast_f32_bf16(const float* __restrict__ in,
                              unsigned short* __restrict__ out, int n) {
  int i = (blockIdx.x * blockDim.x + threadIdx.x) * 8;
  if (i + 8 <= n) {
    v8fv x = *(const v8fv*)(in + i);
    v8bf y;
#pragma unroll
    for (int q = 0; q < 8; ++q) y[q] = (__bf16)x[q];
    *(v8bf*)((__bf16*)out + i) = y;
  }
}

// ---------------------------------------------------------------------------
// Kernel 1b: f32 (KxN) -> bf16 transposed (NxK).  LDS-tiled 32x32.
// One-time weight prep so GEMM B tiles are verbatim async copies.
// ---------------------------------------------------------------------------
__global__ __launch_bounds__(256) void cast_transpose_bf16(
    const float* __restrict__ in, unsigned short* __restrict__ out, int K, int N) {
  __shared__ __bf16 t[32][33];
  const int k0 = blockIdx.y * 32, n0 = blockIdx.x * 32;
  const int tx = threadIdx.x & 31, ty = threadIdx.x >> 5;   // ty: 0..7
#pragma unroll
  for (int r = 0; r < 32; r += 8)
    t[ty + r][tx] = (__bf16)in[(size_t)(k0 + ty + r) * N + n0 + tx];
  __syncthreads();
  __bf16* o = (__bf16*)out;
#pragma unroll
  for (int r = 0; r < 32; r += 8)
    o[(size_t)(n0 + ty + r) * K + k0 + tx] = t[tx][ty + r];
}

// ---------------------------------------------------------------------------
// Kernel 2: bf16 GEMM  C(MxN,f32) = A(MxK) * B^T(NxK); 128x128 tile / block.
// 8 waves: {wm 0..3} x {wn 0..1}; wave owns 32x64 = 8 WMMA accs.
// Both tiles staged via async global->LDS b128, double-buffered:
// issue tile t+1 (4 loads/thread), s_wait_asynccnt 4 drains tile t (in-order).
// ---------------------------------------------------------------------------
__global__ __launch_bounds__(256) void gemm_bf16_kernel(
    const unsigned short* __restrict__ A_, const unsigned short* __restrict__ Bt_,
    float* __restrict__ C, int M, int N, int K) {
  const __bf16* A  = (const __bf16*)A_;
  const __bf16* Bt = (const __bf16*)Bt_;

  __shared__ __bf16 As[2][128 * 32];   // A tile   (128 m x 32 k)   2 x 8KB
  __shared__ __bf16 Bts[2][128 * 32];  // B^T tile (128 n x 32 k)   2 x 8KB

  const int tid  = threadIdx.x;
  const int lane = tid & 31;
  const int wave = tid >> 5;
  const int wm = wave & 3;          // 32-row group
  const int wn = wave >> 2;         // 64-col group (0..1)
  const int m0 = blockIdx.y * 128;
  const int n0 = blockIdx.x * 128;

  v8f acc[2][4];
#pragma unroll
  for (int mi = 0; mi < 2; ++mi)
#pragma unroll
    for (int ni = 0; ni < 4; ++ni) acc[mi][ni] = v8f_zero();

  // stage one 128x32 tile pair into buffer `buf` (4 async b128 / thread)
  auto stage = [&](int buf, int k0) {
#pragma unroll
    for (int t = 0; t < 2; ++t) {
      const int e = (tid + t * 256) * 8;        // bf16 element index
      const int r = e >> 5, c = e & 31;         // row (m or n), k offset
      async_load_b128(&As[buf][e],  &A[(size_t)(m0 + r) * K + k0 + c]);
      async_load_b128(&Bts[buf][e], &Bt[(size_t)(n0 + r) * K + k0 + c]);
    }
  };

  stage(0, 0);
  int buf = 0;
  for (int k0 = 0; k0 < K; k0 += 32) {
    if (k0 + 32 < K) {
      stage(buf ^ 1, k0 + 32);   // prefetch next tile into other buffer
      wait_async4();             // drain current tile only
    } else {
      wait_async0();
    }
    __syncthreads();

    v16bf a0 = load_frag_a(As[buf] + (wm * 32 + 0) * 32, 32);
    v16bf a1 = load_frag_a(As[buf] + (wm * 32 + 16) * 32, 32);
#pragma unroll
    for (int ni = 0; ni < 4; ++ni) {
      v16bf b = load_frag_bT(Bts[buf] + (wn * 64 + ni * 16) * 32, 32);
      acc[0][ni] = __builtin_amdgcn_wmma_f32_16x16x32_bf16(false, a0, false, b, (short)0, acc[0][ni], false, false);
      acc[1][ni] = __builtin_amdgcn_wmma_f32_16x16x32_bf16(false, a1, false, b, (short)0, acc[1][ni], false, false);
    }
    __syncthreads();             // guard buffer reuse by next stage()
    buf ^= 1;
  }

  const int n  = lane & 15;
  const int hf = lane >> 4;
#pragma unroll
  for (int mi = 0; mi < 2; ++mi) {
#pragma unroll
    for (int ni = 0; ni < 4; ++ni) {
#pragma unroll
      for (int r = 0; r < 8; ++r) {
        const size_t row = (size_t)(m0 + wm * 32 + mi * 16 + r + 8 * hf) * N;
        C[row + n0 + wn * 64 + ni * 16 + n] = acc[mi][ni][r];
      }
    }
  }
}

// ---------------------------------------------------------------------------
// Kernel 3: RoPE + split qkv (f32, (B,L,3,H,D)) -> q,k bf16 (B,H,L,D) and
//           v bf16 TRANSPOSED as (B,H,D,L) so P@V B-fragments are contiguous.
// ---------------------------------------------------------------------------
__global__ __launch_bounds__(64) void rope_split_kernel(
    const float* __restrict__ qkv, unsigned short* __restrict__ q_,
    unsigned short* __restrict__ k_, unsigned short* __restrict__ vT_) {
  __bf16* q = (__bf16*)q_;
  __bf16* k = (__bf16*)k_;
  __bf16* vT = (__bf16*)vT_;

  const int d   = threadIdx.x;                 // 0..63
  const int idx = blockIdx.x;                  // (b*H + h)*L + l
  const int l   = idx % L_;
  const int h   = (idx / L_) % H_;
  const int b   = idx / (L_ * H_);

  const size_t rowbase = ((size_t)(b * L_ + l)) * 3 * H_ * D_;
  const float* qp = qkv + rowbase + 0 * H_ * D_ + h * D_;
  const float* kp = qkv + rowbase + 1 * H_ * D_ + h * D_;
  const float* vp = qkv + rowbase + 2 * H_ * D_ + h * D_;

  const float invf = powf(10000.0f, -(float)(2 * (d & 31)) / 64.0f);
  const float ang = (float)l * invf;
  const float c = __cosf(ang), s = __sinf(ang);

  const float qv = qp[d], kv = kp[d];
  const float qrot = (d < 32) ? -qp[d + 32] : qp[d - 32];
  const float krot = (d < 32) ? -kp[d + 32] : kp[d - 32];

  const size_t o = (size_t)idx * D_ + d;
  q[o] = (__bf16)(qv * c + qrot * s);
  k[o] = (__bf16)(kv * c + krot * s);
  vT[((size_t)(b * H_ + h) * D_ + d) * L_ + l] = (__bf16)vp[d];
}

// ---------------------------------------------------------------------------
// Kernel 4: fused flash attention + kerple + cross-head DAPE MLP.
// Grid: (L/16, B). Block: 512 threads = 16 waves; wave h == head h.
// ---------------------------------------------------------------------------
__global__ __launch_bounds__(512) void attn_kernel(
    const unsigned short* __restrict__ qr_, const unsigned short* __restrict__ kr_,
    const unsigned short* __restrict__ vTr_,
    const float* __restrict__ log_p, const float* __restrict__ log_a,
    const float* __restrict__ w1, const float* __restrict__ b1,
    const float* __restrict__ w2, const float* __restrict__ b2,
    unsigned short* __restrict__ obf_) {
  const __bf16* qr = (const __bf16*)qr_;
  const __bf16* kr = (const __bf16*)kr_;
  const __bf16* vT = (const __bf16*)vTr_;
  __bf16* obf = (__bf16*)obf_;

  __shared__ __bf16 sCmb[256][32];     // [pos][feature]: scores | kerple
  __shared__ float  sRef[256][16];     // refined bias per (pos, head)
  __shared__ __bf16 sW1[32][32];
  __shared__ __bf16 sW2[32][16];
  __shared__ float  sB1[32];
  __shared__ float  sB2[16];
  __shared__ __bf16 sP[16][16][32];    // per-wave staging (hdn, then P)

  const int tid  = threadIdx.x;
  const int lane = tid & 31;
  const int h    = tid >> 5;           // wave index == head
  const int n    = lane & 15;          // tile column
  const int hf   = lane >> 4;          // half-wave
  const int it   = blockIdx.x;
  const int b    = blockIdx.y;

  for (int i = tid; i < 32 * 32; i += 512) sW1[i >> 5][i & 31] = (__bf16)w1[i];
  for (int i = tid; i < 32 * 16; i += 512) sW2[i >> 4][i & 15] = (__bf16)w2[i];
  if (tid < 32) sB1[tid] = b1[tid];
  if (tid < 16) sB2[tid] = b2[tid];
  __syncthreads();

  const float p_h = log1pf(__expf(log_p[h]));   // softplus
  const float a_h = log1pf(__expf(log_a[h]));

  const size_t head_base = ((size_t)(b * H_ + h)) * L_ * D_;
  const __bf16* qptr = qr + head_base + (size_t)it * 16 * D_;
  const v16bf aq0 = load_frag_a(qptr, D_);
  const v16bf aq1 = load_frag_a(qptr + 32, D_);

  // loop-invariant MLP weight fragments (hoisted)
  const v16bf bw10 = load_frag_b_rm(&sW1[0][0], 32);
  const v16bf bw11 = load_frag_b_rm(&sW1[0][16], 32);
  const v16bf bw2  = load_frag_b_rm(&sW2[0][0], 16);

  v8f o0 = v8f_zero(), o1 = v8f_zero(), o2 = v8f_zero(), o3 = v8f_zero();
  float m_run[8], l_run[8];
#pragma unroll
  for (int r = 0; r < 8; ++r) { m_run[r] = -1e30f; l_run[r] = 0.0f; }

  for (int jt = 0; jt <= it; ++jt) {
    // ---- phase 1: S_h = (Q K^T) * scale ; deposit scores + kerple --------
    const __bf16* kptr = kr + head_base + (size_t)jt * 16 * D_;
    v16bf bk0 = load_frag_bT(kptr, D_);
    v16bf bk1 = load_frag_bT(kptr + 32, D_);
    v8f s = v8f_zero();
    s = __builtin_amdgcn_wmma_f32_16x16x32_bf16(false, aq0, false, bk0, (short)0, s, false, false);
    s = __builtin_amdgcn_wmma_f32_16x16x32_bf16(false, aq1, false, bk1, (short)0, s, false, false);

    float sv[8], ker[8];
#pragma unroll
    for (int r = 0; r < 8; ++r) {
      const int ii = r + 8 * hf;
      const int ig = it * 16 + ii;
      const int jg = jt * 16 + n;
      sv[r]  = s[r] * 0.125f;                       // 1/sqrt(64)
      ker[r] = -p_h * log1pf(a_h * fabsf((float)(ig - jg)));
      const int pos = ii * 16 + n;
      sCmb[pos][h]      = (__bf16)sv[r];
      sCmb[pos][16 + h] = (__bf16)ker[r];
    }
    __syncthreads();

    // ---- phase 2: DAPE MLP over 256 positions; wave h -> pos [16h,16h+16)
    v16bf ac = load_frag_a(&sCmb[h * 16][0], 32);
    v8f h0 = __builtin_amdgcn_wmma_f32_16x16x32_bf16(false, ac, false, bw10, (short)0, v8f_zero(), false, false);
    v8f h1 = __builtin_amdgcn_wmma_f32_16x16x32_bf16(false, ac, false, bw11, (short)0, v8f_zero(), false, false);
#pragma unroll
    for (int r = 0; r < 8; ++r) {
      float x0 = h0[r] + sB1[n];
      float x1 = h1[r] + sB1[16 + n];
      x0 = 0.5f * x0 * (1.0f + erff(x0 * 0.70710678f));   // exact gelu
      x1 = 0.5f * x1 * (1.0f + erff(x1 * 0.70710678f));
      const int m = r + 8 * hf;
      sP[h][m][n]      = (__bf16)x0;
      sP[h][m][16 + n] = (__bf16)x1;
    }
    v16bf ah = load_frag_a(&sP[h][0][0], 32);
    v8f rf = __builtin_amdgcn_wmma_f32_16x16x32_bf16(false, ah, false, bw2, (short)0, v8f_zero(), false, false);
#pragma unroll
    for (int r = 0; r < 8; ++r)
      sRef[h * 16 + r + 8 * hf][n] = rf[r] + sB2[n];
    __syncthreads();

    // ---- phase 3: bias add + causal mask + online softmax + P@V ----------
    float oscale[8];
#pragma unroll
    for (int r = 0; r < 8; ++r) {
      const int ii = r + 8 * hf;
      const int ig = it * 16 + ii;
      const int jg = jt * 16 + n;
      float x = sv[r] + ker[r] + sRef[ii * 16 + n][h];
      if (jg > ig) x = -1e30f;

      float rmax = x;
#pragma unroll
      for (int off = 1; off < 16; off <<= 1)
        rmax = fmaxf(rmax, __shfl_xor(rmax, off, 32));
      const float newM = fmaxf(m_run[r], rmax);
      const float pv = __expf(x - newM);
      float rsum = pv;
#pragma unroll
      for (int off = 1; off < 16; off <<= 1)
        rsum += __shfl_xor(rsum, off, 32);
      const float co = __expf(m_run[r] - newM);
      l_run[r] = l_run[r] * co + rsum;
      m_run[r] = newM;
      oscale[r] = co;
      sP[h][ii][n]      = (__bf16)pv;
      sP[h][ii][16 + n] = (__bf16)0.0f;
    }
#pragma unroll
    for (int r = 0; r < 8; ++r) {
      o0[r] *= oscale[r]; o1[r] *= oscale[r];
      o2[r] *= oscale[r]; o3[r] *= oscale[r];
    }
    v16bf ap = load_frag_a(&sP[h][0][0], 32);
    // V^T fragments: contiguous 32B per lane from (B,H,D,L) layout
    const __bf16* vTh = vT + (size_t)(b * H_ + h) * D_ * L_ + jt * 16;
    v16bf bv0 = load_frag_bT_k16(vTh + (size_t)(0 * 16) * L_, L_);
    v16bf bv1 = load_frag_bT_k16(vTh + (size_t)(1 * 16) * L_, L_);
    v16bf bv2 = load_frag_bT_k16(vTh + (size_t)(2 * 16) * L_, L_);
    v16bf bv3 = load_frag_bT_k16(vTh + (size_t)(3 * 16) * L_, L_);
    o0 = __builtin_amdgcn_wmma_f32_16x16x32_bf16(false, ap, false, bv0, (short)0, o0, false, false);
    o1 = __builtin_amdgcn_wmma_f32_16x16x32_bf16(false, ap, false, bv1, (short)0, o1, false, false);
    o2 = __builtin_amdgcn_wmma_f32_16x16x32_bf16(false, ap, false, bv2, (short)0, o2, false, false);
    o3 = __builtin_amdgcn_wmma_f32_16x16x32_bf16(false, ap, false, bv3, (short)0, o3, false, false);
  }

  // ---- finalize: O /= l; store as (B, L, H*D) bf16 for the out-proj GEMM
#pragma unroll
  for (int r = 0; r < 8; ++r) {
    const int ii = r + 8 * hf;
    const float inv = 1.0f / l_run[r];
    const size_t row = (size_t)(b * L_ + it * 16 + ii) * (H_ * D_) + h * D_;
    obf[row + 0  + n] = (__bf16)(o0[r] * inv);
    obf[row + 16 + n] = (__bf16)(o1[r] * inv);
    obf[row + 32 + n] = (__bf16)(o2[r] * inv);
    obf[row + 48 + n] = (__bf16)(o3[r] * inv);
  }
}

// ---------------------------------------------------------------------------
// Host-side orchestration
// ---------------------------------------------------------------------------
extern "C" void kernel_launch(void* const* d_in, const int* in_sizes, int n_in,
                              void* d_out, int out_size, void* d_ws, size_t ws_size,
                              hipStream_t stream) {
  (void)in_sizes; (void)n_in; (void)out_size; (void)ws_size;

  const float* x     = (const float*)d_in[0];
  const float* w_qkv = (const float*)d_in[1];
  const float* w_o   = (const float*)d_in[2];
  const float* log_p = (const float*)d_in[3];
  const float* log_a = (const float*)d_in[4];
  const float* w1    = (const float*)d_in[5];
  const float* b1    = (const float*)d_in[6];
  const float* w2    = (const float*)d_in[7];
  const float* b2    = (const float*)d_in[8];
  float* out = (float*)d_out;

  const int BL = B_ * L_;               // 2048
  const int N_QKV = 3 * H_ * D_;        // 3072

  char* ws = (char*)d_ws;
  unsigned short* xb   = (unsigned short*)ws;  ws += (size_t)BL * HID_ * 2;
  unsigned short* wqbT = (unsigned short*)ws;  ws += (size_t)HID_ * N_QKV * 2;   // (3072 x 1024)
  unsigned short* wobT = (unsigned short*)ws;  ws += (size_t)(H_ * D_) * HID_ * 2; // (1024 x 1024)
  float*          qkv  = (float*)ws;           ws += (size_t)BL * N_QKV * 4;
  unsigned short* qr   = (unsigned short*)ws;  ws += (size_t)B_ * H_ * L_ * D_ * 2;
  unsigned short* kr   = (unsigned short*)ws;  ws += (size_t)B_ * H_ * L_ * D_ * 2;
  unsigned short* vtr  = (unsigned short*)ws;  ws += (size_t)B_ * H_ * L_ * D_ * 2;
  unsigned short* obf  = (unsigned short*)ws;  ws += (size_t)BL * (H_ * D_) * 2;

  // 1) activation cast + one-time weight cast/transpose to (N x K) bf16
  {
    int n = BL * HID_;
    cast_f32_bf16<<<(n / 8 + 255) / 256, 256, 0, stream>>>(x, xb, n);
    cast_transpose_bf16<<<dim3(N_QKV / 32, HID_ / 32), 256, 0, stream>>>(
        w_qkv, wqbT, HID_, N_QKV);
    cast_transpose_bf16<<<dim3(HID_ / 32, (H_ * D_) / 32), 256, 0, stream>>>(
        w_o, wobT, H_ * D_, HID_);
  }

  // 2) qkv = x @ w_qkv   (2048 x 3072, K=1024), 128x128 tiles, async dbl-buf
  gemm_bf16_kernel<<<dim3(N_QKV / 128, BL / 128), 256, 0, stream>>>(
      xb, wqbT, qkv, BL, N_QKV, HID_);

  // 3) RoPE + split into q,k (B,H,L,D) and v^T (B,H,D,L)
  rope_split_kernel<<<B_ * H_ * L_, 64, 0, stream>>>(qkv, qr, kr, vtr);

  // 4) fused attention (flash + kerple + DAPE MLP)
  attn_kernel<<<dim3(L_ / 16, B_), 512, 0, stream>>>(
      qr, kr, vtr, log_p, log_a, w1, b1, w2, b2, obf);

  // 5) out = attn_out @ w_o   (2048 x 1024, K=1024)
  gemm_bf16_kernel<<<dim3(HID_ / 128, BL / 128), 256, 0, stream>>>(
      obf, wobT, out, BL, HID_, HID_);
}
